// Attention_2783138808527
// MI455X (gfx1250) — matrix operand order repaired and checked by reading, not compiled, not run on hardware
//
#include <hip/hip_runtime.h>
#include <hip/hip_bf16.h>

typedef __attribute__((ext_vector_type(16))) __bf16 v16bf;
typedef __attribute__((ext_vector_type(8)))  __bf16 v8bf;
typedef __attribute__((ext_vector_type(4)))  __bf16 v4bf;
typedef __attribute__((ext_vector_type(16))) float  v16f;
typedef __attribute__((ext_vector_type(8)))  float  v8f;
typedef __attribute__((ext_vector_type(4)))  float  v4f;

#define B_ 64
#define S_ 1024
#define H_ 1024
#define STILE 128
#define NWAVES 8

// Native CDNA5 V_TANH_F32 if the toolchain exposes it; OCML fallback otherwise.
__device__ __forceinline__ float fast_tanh(float x) {
#if __has_builtin(__builtin_amdgcn_tanhf)
  return __builtin_amdgcn_tanhf(x);
#elif __has_builtin(__builtin_amdgcn_tanh_f32)
  return __builtin_amdgcn_tanh_f32(x);
#else
  return tanhf(x);
#endif
}

// ---------------------------------------------------------------------------
// Kernel 1: hb[b][o] = attn_b[o] + sum_k hidden[b][k] * attn_W[o][k]
// ---------------------------------------------------------------------------
__global__ __launch_bounds__(256) void hproj_kernel(
    const float* __restrict__ hidden, const float* __restrict__ attn_W,
    const float* __restrict__ attn_b, float* __restrict__ hb) {
  int gid = blockIdx.x * 256 + threadIdx.x;   // gid = b*1024 + o
  int o = gid & (H_ - 1);
  int b = gid >> 10;
  const v4f* wp = (const v4f*)(attn_W + (size_t)o * (2 * H_));
  const v4f* hp = (const v4f*)(hidden + (size_t)b * H_);
  float acc = 0.f;
#pragma unroll 4
  for (int k = 0; k < H_ / 4; ++k) {
    v4f w = wp[k], h = hp[k];
    acc += w.x * h.x + w.y * h.y + w.z * h.z + w.w * h.w;
  }
  hb[gid] = acc + attn_b[o];
}

// ---------------------------------------------------------------------------
// Kernel 2: per (b, 128-row s-tile): stage enc tile as bf16 in LDS, then
// WMMA-GEMM against We^T with fused tanh(.)·v_W reduction over o.
// ---------------------------------------------------------------------------
__global__ __launch_bounds__(256) void escore_kernel(
    const float* __restrict__ enc,     // (S,B,H) f32
    const float* __restrict__ attn_W,  // (H,2H) f32, We = cols [H,2H)
    const float* __restrict__ hb,      // (B,H)  h_proj + attn_b
    const float* __restrict__ vW,      // (H)
    float* __restrict__ scores)        // (B,S) raw scores out
{
  __shared__ __bf16 Abuf[STILE * H_];  // 256 KB bf16 enc tile
  __shared__ float  sred[STILE];       // per-s score accumulators

  const int tid    = threadIdx.x;
  const int wave   = tid >> 5;
  const int lane   = tid & 31;
  const int lane15 = lane & 15;
  const int khalf  = lane >> 4;

  const int b  = blockIdx.x >> 3;   // 64 batches
  const int s0 = (blockIdx.x & 7) * STILE;

  if (tid < STILE) sred[tid] = 0.f;

  // ---- stage: enc[s0..s0+127, b, :] -> bf16 LDS (row-major s x h) ----
  {
    const v4f* src = (const v4f*)enc;
#pragma unroll 4
    for (int it = 0; it < (STILE * (H_ / 4)) / 256; ++it) {
      int c  = tid + it * 256;
      int r  = c >> 8;        // / (H_/4)
      int h4 = c & 255;
      v4f v = src[(((size_t)(s0 + r) * B_ + b) * H_ >> 2) + h4];
      v4bf bv = __builtin_convertvector(v, v4bf);
      *(v4bf*)(&Abuf[r * H_ + h4 * 4]) = bv;
    }
  }
  __syncthreads();

  // ---- main GEMM: each wave owns a 16-wide o strip per outer iteration ----
  for (int nouter = 0; nouter < H_; nouter += NWAVES * 16) {
    const int n0 = nouter + wave * 16;
    const int oc = n0 + lane15;                       // this lane's o column
    const float* wrow = attn_W + (size_t)oc * (2 * H_) + H_;  // We row oc

    v8f acc[8];
#pragma unroll
    for (int m = 0; m < 8; ++m) acc[m] = (v8f){0, 0, 0, 0, 0, 0, 0, 0};

#pragma unroll 2
    for (int kb = 0; kb < H_ / 32; ++kb) {
      // B fragment: K = kb*32 + khalf*16 + j  (16 contiguous f32 -> bf16)
      const v4f* bp = (const v4f*)(wrow + kb * 32 + khalf * 16);
      v4f f0 = bp[0], f1 = bp[1], f2 = bp[2], f3 = bp[3];
      v8f  lo = __builtin_shufflevector(f0, f1, 0, 1, 2, 3, 4, 5, 6, 7);
      v8f  hi = __builtin_shufflevector(f2, f3, 0, 1, 2, 3, 4, 5, 6, 7);
      v16f ff = __builtin_shufflevector(lo, hi, 0, 1, 2, 3, 4, 5, 6, 7, 8, 9,
                                        10, 11, 12, 13, 14, 15);
      v16bf bfrag = __builtin_convertvector(ff, v16bf);

#pragma unroll
      for (int m = 0; m < 8; ++m) {
        // A fragment: row M = m*16 + lane15, K = kb*32 + khalf*16 + j
        const __bf16* ap = &Abuf[(m * 16 + lane15) * H_ + kb * 32 + khalf * 16];
        v8bf a0 = *(const v8bf*)ap;
        v8bf a1 = *(const v8bf*)(ap + 8);
        v16bf afrag = __builtin_shufflevector(a0, a1, 0, 1, 2, 3, 4, 5, 6, 7,
                                              8, 9, 10, 11, 12, 13, 14, 15);
        acc[m] = __builtin_amdgcn_wmma_f32_16x16x32_bf16(
            false, afrag, false, bfrag, (short)0, acc[m], false, false);
      }
    }

    // ---- fused epilogue: tanh(e + hb) * vW, reduce over o (lanes) ----
    const float hbl = hb[(size_t)b * H_ + oc];
    const float vwl = vW[oc];
#pragma unroll
    for (int m = 0; m < 8; ++m) {
#pragma unroll
      for (int i = 0; i < 8; ++i) {
        float e = fast_tanh(acc[m][i] + hbl) * vwl;
        e += __shfl_xor(e, 8);
        e += __shfl_xor(e, 4);
        e += __shfl_xor(e, 2);
        e += __shfl_xor(e, 1);
        if (lane15 == 0) {
          atomicAdd(&sred[m * 16 + khalf * 8 + i], e);
        }
      }
    }
  }

  __syncthreads();
  if (tid < STILE) scores[(size_t)b * S_ + s0 + tid] = sred[tid];
}

// ---------------------------------------------------------------------------
// Kernel 3: mask + softmax over s, in place on d_out. One block per batch.
// ---------------------------------------------------------------------------
__global__ __launch_bounds__(256) void softmax_kernel(
    const int* __restrict__ mask, float* __restrict__ out) {
  __shared__ float red[256];
  const int b = blockIdx.x;
  const int tid = threadIdx.x;

  float v[4];
  float mx = -3.402823466e38f;
#pragma unroll
  for (int k = 0; k < 4; ++k) {
    int s = tid + k * 256;
    float sc = out[(size_t)b * S_ + s];
    if (mask[(size_t)b * S_ + s] == 0) sc = -1.0e10f;
    v[k] = sc;
    mx = fmaxf(mx, sc);
  }
  red[tid] = mx;
  __syncthreads();
  for (int off = 128; off > 0; off >>= 1) {
    if (tid < off) red[tid] = fmaxf(red[tid], red[tid + off]);
    __syncthreads();
  }
  mx = red[0];
  __syncthreads();

  float sum = 0.f;
#pragma unroll
  for (int k = 0; k < 4; ++k) {
    v[k] = expf(v[k] - mx);
    sum += v[k];
  }
  red[tid] = sum;
  __syncthreads();
  for (int off = 128; off > 0; off >>= 1) {
    if (tid < off) red[tid] += red[tid + off];
    __syncthreads();
  }
  const float inv = 1.f / red[0];
#pragma unroll
  for (int k = 0; k < 4; ++k)
    out[(size_t)b * S_ + tid + k * 256] = v[k] * inv;
}

// ---------------------------------------------------------------------------
extern "C" void kernel_launch(void* const* d_in, const int* in_sizes, int n_in,
                              void* d_out, int out_size, void* d_ws,
                              size_t ws_size, hipStream_t stream) {
  const float* hidden = (const float*)d_in[0];
  const float* enc    = (const float*)d_in[1];
  const int*   mask   = (const int*)d_in[2];
  const float* attn_W = (const float*)d_in[3];
  const float* attn_b = (const float*)d_in[4];
  const float* vW     = (const float*)d_in[5];
  float* out = (float*)d_out;
  float* hb  = (float*)d_ws;  // 64*1024 f32 = 256 KB scratch

  hproj_kernel<<<(B_ * H_) / 256, 256, 0, stream>>>(hidden, attn_W, attn_b, hb);
  escore_kernel<<<B_ * (S_ / STILE), 256, 0, stream>>>(enc, attn_W, hb, vW, out);
  softmax_kernel<<<B_, 256, 0, stream>>>(mask, out);
}
